// MultiHeadAttention_23527830847514
// MI455X (gfx1250) — matrix operand order repaired and checked by reading, not compiled, not run on hardware
//
#include <hip/hip_runtime.h>

// ---------------------------------------------------------------------------
// MHA on gfx1250 (MI455X).
// cvt(fp32->bf16) -> 3x bf16 WMMA GEMM (Q/K/V proj, Q pre-scaled by 1/8)
//   -> bf16 flash attention (64-key blocks) -> bf16 WMMA GEMM (O proj, fp32).
// GEMM: 128x256 block tile, 64x64 per wave (16 WMMA/k-step), double-buffered
// LDS filled with global_load_async_to_lds_b128 (ASYNCcnt) overlapped with
// compute. No fp32<->bf16 conversion inside any K-loop.
//
// Workspace (64 MB as ushort elements):
//   Xq|Xk|Xv (4M each) | Wq|Wk|Wv|Wo (1M each) | Qbf|Kbf|Vbf (4M) | Abf (4M)
// ---------------------------------------------------------------------------

typedef __attribute__((ext_vector_type(16))) __bf16        v16bf;
typedef __attribute__((ext_vector_type(8)))  float         v8f;
typedef __attribute__((ext_vector_type(4)))  float         v4f;
typedef __attribute__((ext_vector_type(4)))  unsigned int  v4u;

union FragBF {
    v16bf          v;     // 32 bytes: 16 bf16 elements (8 VGPRs)
    v4u            u[2];  // two 16-byte chunks
    unsigned short s[16];
};

// float -> bf16 bits, round-to-nearest-even (used only outside hot loops)
__device__ __forceinline__ unsigned short f2bf(float f) {
    unsigned u = __builtin_bit_cast(unsigned, f);
    u += 0x7FFFu + ((u >> 16) & 1u);
    return (unsigned short)(u >> 16);
}

// LDS byte offset of a __shared__ object: low 32 bits of the flat address
__device__ __forceinline__ unsigned lds_off(const void* p) {
    return (unsigned)(unsigned long long)p;
}

// 32B async copy global->LDS (two b128 ops sharing one address pair)
__device__ __forceinline__ void async_cp32(unsigned ldsoff, unsigned long long g) {
    asm volatile(
        "global_load_async_to_lds_b128 %0, %1, off\n\t"
        "global_load_async_to_lds_b128 %0, %1, off offset:16"
        :: "v"(ldsoff), "v"(g) : "memory");
}

// ===========================================================================
// One-time fp32 -> bf16 conversion (n8 = n/8 vector groups)
// ===========================================================================
__global__ __launch_bounds__(256)
void cvt_f32_bf16(const float* __restrict__ src, unsigned short* __restrict__ dst,
                  int n8) {
    const int i = blockIdx.x * 256 + threadIdx.x;
    if (i >= n8) return;
    v4f a = *(const v4f*)(src + (size_t)i * 8);
    v4f b = *(const v4f*)(src + (size_t)i * 8 + 4);
    union { v4u u; unsigned short s[8]; } o;
#pragma unroll
    for (int e = 0; e < 4; ++e) {
        o.s[e]     = f2bf(a[e]);
        o.s[4 + e] = f2bf(b[e]);
    }
    *(v4u*)(dst + (size_t)i * 8) = o.u;
}

// ===========================================================================
// GEMM: C = (A[M,K] @ W[N,K]^T + bias[N]) * scale   (A,W bf16)
// Writes Cb (bf16) and/or Cf (fp32); either may be null.
// Block 256 thr (8 waves, 2x4), tile 128(M)x256(N), K-step 32, wave = 64x64.
// Double-buffered async staging overlapped with WMMA.
// ===========================================================================
__global__ __launch_bounds__(256)
void gemm_bf16(const unsigned short* __restrict__ A,
               const unsigned short* __restrict__ W,
               const float* __restrict__ bias,
               unsigned short* __restrict__ Cb, float* __restrict__ Cf,
               int M, int N, int K, float scale) {
    // stride 40 halfs = 80B: 16B-aligned chunks, spread banks
    __shared__ alignas(16) unsigned short lsA[2][128][40];
    __shared__ alignas(16) unsigned short lsB[2][256][40];

    const int tid  = threadIdx.x;
    const int wave = tid >> 5;
    const int lane = tid & 31;
    const int l16  = lane & 15;
    const int half = lane >> 4;

    const int blockM = blockIdx.y * 128;
    const int blockN = blockIdx.x * 256;
    const int waveM  = (wave & 1) * 64;
    const int waveN  = (wave >> 1) * 64;

    v8f acc[4][4] = {};

    // staging map: thread t copies 32B halves of row r for A, W row r, W row r+128
    const int r  = tid >> 1;
    const int c0 = (tid & 1) * 32;  // byte offset within the 64B bf16 row
    const unsigned short* agp  = A + (size_t)(blockM + r) * K;
    const unsigned short* bgp0 = W + (size_t)(blockN + r) * K;
    const unsigned short* bgp1 = W + (size_t)(blockN + r + 128) * K;
    const unsigned aoff[2]  = { lds_off(&lsA[0][r][0]) + c0,
                                lds_off(&lsA[1][r][0]) + c0 };
    const unsigned boff[2]  = { lds_off(&lsB[0][r][0]) + c0,
                                lds_off(&lsB[1][r][0]) + c0 };
    const unsigned boff2[2] = { lds_off(&lsB[0][r + 128][0]) + c0,
                                lds_off(&lsB[1][r + 128][0]) + c0 };

    auto stage = [&](int buf, int k0) {
        async_cp32(aoff[buf],  (unsigned long long)(const void*)(agp  + k0) + c0);
        async_cp32(boff[buf],  (unsigned long long)(const void*)(bgp0 + k0) + c0);
        async_cp32(boff2[buf], (unsigned long long)(const void*)(bgp1 + k0) + c0);
    };

    stage(0, 0);  // prologue fill of buffer 0

    int buf = 0;
    for (int k0 = 0; k0 < K; k0 += 32, buf ^= 1) {
        // own async writes done, then everyone's (they all waited before signaling)
        asm volatile("s_wait_asynccnt 0" ::: "memory");
        __syncthreads();
        if (k0 + 32 < K) stage(buf ^ 1, k0 + 32);  // overlap next tile with compute

        // A fragment (16x32): e0..7 -> K=half*8..+7 ; e8..15 -> K=16+half*8..+7
        FragBF afrag[4];
#pragma unroll
        for (int i = 0; i < 4; ++i) {
            const int rr = waveM + i * 16 + l16;
            afrag[i].u[0] = *(const v4u*)&lsA[buf][rr][half * 8];
            afrag[i].u[1] = *(const v4u*)&lsA[buf][rr][16 + half * 8];
        }
        // B fragment (32x16): lane=N, e0..7 -> K=half*8..+7 ; e8..15 -> K=16+half*8..+7
        FragBF bfrag[4];
#pragma unroll
        for (int j = 0; j < 4; ++j) {
            const int rr = waveN + j * 16 + l16;
            bfrag[j].u[0] = *(const v4u*)&lsB[buf][rr][half * 8];
            bfrag[j].u[1] = *(const v4u*)&lsB[buf][rr][16 + half * 8];
        }
#pragma unroll
        for (int i = 0; i < 4; ++i)
#pragma unroll
            for (int j = 0; j < 4; ++j)
                acc[i][j] = __builtin_amdgcn_wmma_f32_16x16x32_bf16(
                    false, afrag[i].v, false, bfrag[j].v,
                    (short)0, acc[i][j], false, false);
    }

    // epilogue: C layout VGPR v -> row v + half*8, col = lane N
#pragma unroll
    for (int j = 0; j < 4; ++j) {
        const int col = blockN + waveN + j * 16 + l16;
        const float b = bias[col];
#pragma unroll
        for (int i = 0; i < 4; ++i) {
            const int rbase = blockM + waveM + i * 16 + half * 8;
#pragma unroll
            for (int v = 0; v < 8; ++v) {
                const float r2 = (acc[i][j][v] + b) * scale;
                const size_t idx = (size_t)(rbase + v) * N + col;
                if (Cf) Cf[idx] = r2;
                if (Cb) Cb[idx] = f2bf(r2);
            }
        }
    }
}

// ===========================================================================
// Flash attention over bf16 Q/K/V (Q pre-scaled by 1/sqrt(dk)).
// Grid (S/64, H), block 128 thr (4 waves); wave owns 16 q rows.
// 64-key blocks: 16 WMMA per iteration, softmax/shuffle/barrier overhead
// amortized over twice the keys.
// ===========================================================================
#define S_LEN 4096
#define DM    1024
#define HD    64

__global__ __launch_bounds__(128)
void flash_attn(const unsigned short* __restrict__ Qp,
                const unsigned short* __restrict__ Kp,
                const unsigned short* __restrict__ Vp,
                unsigned short* __restrict__ Op) {
    __shared__ alignas(16) unsigned short lsVT[64][72];    // [d][key0..63] V^T
    __shared__ alignas(16) unsigned short lsP[4][16][72];  // per-wave P (16x64)

    const int tid  = threadIdx.x;
    const int wave = tid >> 5;
    const int lane = tid & 31;
    const int l16  = lane & 15;
    const int half = lane >> 4;

    const int h    = blockIdx.y;
    const int q0   = blockIdx.x * 64 + wave * 16;
    const int hcol = h * HD;

    // Q A-fragments: straight bf16 b128 loads (already scaled)
    FragBF qa[2];
    {
        const unsigned short* qrow = Qp + (size_t)(q0 + l16) * DM + hcol;
#pragma unroll
        for (int c = 0; c < 2; ++c) {
            qa[c].u[0] = *(const v4u*)(qrow + c * 32 + half * 8);
            qa[c].u[1] = *(const v4u*)(qrow + c * 32 + 16 + half * 8);
        }
    }

    float mrow[8], lrow[8];
#pragma unroll
    for (int v = 0; v < 8; ++v) { mrow[v] = -1e30f; lrow[v] = 0.0f; }
    v8f oacc[4] = {};

    for (int kb = 0; kb < S_LEN; kb += 64) {
        __syncthreads();
        // stage V^T tile: lsVT[d][key] = V[kb+key][hcol+d]  (byte shuffle only)
        {
            const int k  = tid >> 1;          // 0..63
            const int dq = (tid & 1) * 32;    // 0 or 32
            const unsigned short* vp = Vp + (size_t)(kb + k) * DM + hcol + dq;
            FragBF t0, t1;
            t0.u[0] = *(const v4u*)vp;
            t0.u[1] = *(const v4u*)(vp + 8);
            t1.u[0] = *(const v4u*)(vp + 16);
            t1.u[1] = *(const v4u*)(vp + 24);
#pragma unroll
            for (int e = 0; e < 16; ++e) {
                lsVT[dq + e][k]      = t0.s[e];
                lsVT[dq + 16 + e][k] = t1.s[e];
            }
        }
        __syncthreads();

        // scores (16q x 64k): K B-fragments straight from global bf16
        // (lane = key, e0..7 -> d = half*8 + e ; e8..15 -> d = 16 + half*8 + e-8)
        v8f sc[4];
#pragma unroll
        for (int nt = 0; nt < 4; ++nt) {
            v8f s = {};
#pragma unroll
            for (int c = 0; c < 2; ++c) {
                FragBF kf;
                const unsigned short* kp =
                    Kp + (size_t)(kb + nt * 16 + l16) * DM + hcol + c * 32 + half * 8;
                kf.u[0] = *(const v4u*)kp;
                kf.u[1] = *(const v4u*)(kp + 16);
                s = __builtin_amdgcn_wmma_f32_16x16x32_bf16(
                    false, qa[c].v, false, kf.v, (short)0, s, false, false);
            }
            sc[nt] = s;
        }

        // online softmax (row = v + half*8; reductions stay intra-half)
        float alpha[8];
#pragma unroll
        for (int v = 0; v < 8; ++v) {
            float bm = fmaxf(fmaxf(sc[0][v], sc[1][v]), fmaxf(sc[2][v], sc[3][v]));
            bm = fmaxf(bm, __shfl_xor(bm, 1));
            bm = fmaxf(bm, __shfl_xor(bm, 2));
            bm = fmaxf(bm, __shfl_xor(bm, 4));
            bm = fmaxf(bm, __shfl_xor(bm, 8));
            const float mn = fmaxf(mrow[v], bm);
            alpha[v] = __expf(mrow[v] - mn);
            mrow[v]  = mn;
#pragma unroll
            for (int nt = 0; nt < 4; ++nt)
                sc[nt][v] = __expf(sc[nt][v] - mn);
            float rs = (sc[0][v] + sc[1][v]) + (sc[2][v] + sc[3][v]);
            rs += __shfl_xor(rs, 1);
            rs += __shfl_xor(rs, 2);
            rs += __shfl_xor(rs, 4);
            rs += __shfl_xor(rs, 8);
            lrow[v] = lrow[v] * alpha[v] + rs;
        }
#pragma unroll
        for (int j = 0; j < 4; ++j)
#pragma unroll
            for (int v = 0; v < 8; ++v)
                oacc[j][v] *= alpha[v];

        // P (C-layout, 16x64) -> LDS -> reload as two 16x32 A-fragments
#pragma unroll
        for (int nt = 0; nt < 4; ++nt)
#pragma unroll
            for (int v = 0; v < 8; ++v)
                lsP[wave][v + half * 8][nt * 16 + l16] = f2bf(sc[nt][v]);
        asm volatile("s_wait_dscnt 0" ::: "memory");

        FragBF pa[2];
#pragma unroll
        for (int c = 0; c < 2; ++c) {
            pa[c].u[0] = *(const v4u*)&lsP[wave][l16][c * 32 + half * 8];
            pa[c].u[1] = *(const v4u*)&lsP[wave][l16][c * 32 + 16 + half * 8];
        }

        // O += P @ V : per key-chunk c, B-fragment lane = d (lsVT row),
        // e0..7 -> key = c*32 + half*8 + e ; e8..15 -> key = c*32 + 16 + half*8 + e-8
#pragma unroll
        for (int c = 0; c < 2; ++c)
#pragma unroll
            for (int j = 0; j < 4; ++j) {
                FragBF vf;
                vf.u[0] = *(const v4u*)&lsVT[j * 16 + l16][c * 32 + half * 8];
                vf.u[1] = *(const v4u*)&lsVT[j * 16 + l16][c * 32 + 16 + half * 8];
                oacc[j] = __builtin_amdgcn_wmma_f32_16x16x32_bf16(
                    false, pa[c].v, false, vf.v, (short)0, oacc[j], false, false);
            }
    }

    // epilogue: normalize, write bf16 for the O-projection GEMM
#pragma unroll
    for (int v = 0; v < 8; ++v) {
        const float inv = 1.0f / lrow[v];
#pragma unroll
        for (int j = 0; j < 4; ++j)
            Op[(size_t)(q0 + v + half * 8) * DM + hcol + j * 16 + l16] =
                f2bf(oacc[j][v] * inv);
    }
}

// ===========================================================================
extern "C" void kernel_launch(void* const* d_in, const int* in_sizes, int n_in,
                              void* d_out, int out_size, void* d_ws, size_t ws_size,
                              hipStream_t stream) {
    const float* query = (const float*)d_in[0];
    const float* key   = (const float*)d_in[1];
    const float* value = (const float*)d_in[2];
    const float* W_q   = (const float*)d_in[3];
    const float* b_q   = (const float*)d_in[4];
    const float* W_k   = (const float*)d_in[5];
    const float* b_k   = (const float*)d_in[6];
    const float* W_v   = (const float*)d_in[7];
    const float* b_v   = (const float*)d_in[8];
    const float* W_o   = (const float*)d_in[9];
    const float* b_o   = (const float*)d_in[10];
    float* out = (float*)d_out;

    const int S = 4096, D = 1024;
    const size_t mat  = (size_t)S * D;   // 4M elements
    const size_t wmat = (size_t)D * D;   // 1M elements

    unsigned short* ws = (unsigned short*)d_ws;   // 64MB total as bf16
    unsigned short* Xq = ws;
    unsigned short* Xk = Xq + mat;
    unsigned short* Xv = Xk + mat;
    unsigned short* Wq = Xv + mat;
    unsigned short* Wk = Wq + wmat;
    unsigned short* Wv = Wk + wmat;
    unsigned short* Wo = Wv + wmat;
    unsigned short* Qb = Wo + wmat;
    unsigned short* Kb = Qb + mat;
    unsigned short* Vb = Kb + mat;
    unsigned short* Ab = Vb + mat;

    // one-time conversions
    const int mat8  = (int)(mat / 8), wmat8 = (int)(wmat / 8);
    cvt_f32_bf16<<<mat8 / 256, 256, 0, stream>>>(query, Xq, mat8);
    cvt_f32_bf16<<<mat8 / 256, 256, 0, stream>>>(key,   Xk, mat8);
    cvt_f32_bf16<<<mat8 / 256, 256, 0, stream>>>(value, Xv, mat8);
    cvt_f32_bf16<<<wmat8 / 256, 256, 0, stream>>>(W_q, Wq, wmat8);
    cvt_f32_bf16<<<wmat8 / 256, 256, 0, stream>>>(W_k, Wk, wmat8);
    cvt_f32_bf16<<<wmat8 / 256, 256, 0, stream>>>(W_v, Wv, wmat8);
    cvt_f32_bf16<<<wmat8 / 256, 256, 0, stream>>>(W_o, Wo, wmat8);

    dim3 ggrid(D / 256, S / 128);  // (4, 32)
    // Q projection pre-scaled by 1/sqrt(d_k) = 0.125
    gemm_bf16<<<ggrid, 256, 0, stream>>>(Xq, Wq, b_q, Qb, nullptr, S, D, D, 0.125f);
    gemm_bf16<<<ggrid, 256, 0, stream>>>(Xk, Wk, b_k, Kb, nullptr, S, D, D, 1.0f);
    gemm_bf16<<<ggrid, 256, 0, stream>>>(Xv, Wv, b_v, Vb, nullptr, S, D, D, 1.0f);
    flash_attn<<<dim3(S / 64, 16), 128, 0, stream>>>(Qb, Kb, Vb, Ab);
    gemm_bf16<<<ggrid, 256, 0, stream>>>(Ab, Wo, b_o, nullptr, out, S, D, D, 1.0f);
}
